// GCN31_13443247637083
// MI455X (gfx1250) — matrix operand order repaired and checked by reading, not compiled, not run on hardware
//
#include <hip/hip_runtime.h>
#include <hip/hip_bf16.h>

typedef __attribute__((ext_vector_type(16))) _Float16 v16h;
typedef __attribute__((ext_vector_type(8)))  _Float16 v8h;
typedef __attribute__((ext_vector_type(8)))  float    v8f;

#define NEG_SLOPE 0.2f

__device__ __forceinline__ float lrelu(float v) { return fmaxf(v, NEG_SLOPE * v); }

// float atomic max via signed/unsigned integer atomics (order-isomorphic bit trick).
__device__ __forceinline__ void atomicMaxF(float* addr, float v) {
    if (v >= 0.0f) atomicMax((int*)addr, __float_as_int(v));
    else           atomicMin((unsigned int*)addr, __float_as_uint(v));
}

// 8 consecutive f32 -> packed v8h (2x global_load_b128 + 4x v_cvt_pk_f16_f32)
__device__ __forceinline__ v8h cvt8(const float* __restrict__ p) {
    float4 f0 = *(const float4*)(p);
    float4 f1 = *(const float4*)(p + 4);
    v8h r;
    r[0] = (_Float16)f0.x; r[1] = (_Float16)f0.y;
    r[2] = (_Float16)f0.z; r[3] = (_Float16)f0.w;
    r[4] = (_Float16)f1.x; r[5] = (_Float16)f1.y;
    r[6] = (_Float16)f1.z; r[7] = (_Float16)f1.w;
    return r;
}

// ---------------------------------------------------------------------------
// Pack per-layer weights into f16, column-major [16 cols][Kpad], zero-padded.
// Virtual K ordering matches concat([xa, x]): rows 0..Ktot-1 of W directly.
// ---------------------------------------------------------------------------
__global__ void prep_weights(const float* __restrict__ W1, const float* __restrict__ W2,
                             const float* __restrict__ W3, const float* __restrict__ W4,
                             _Float16* __restrict__ Wt1, _Float16* __restrict__ Wt2,
                             _Float16* __restrict__ Wt3, _Float16* __restrict__ Wt4) {
    int t = threadIdx.x;
    for (int i = t; i < 16 * 128; i += blockDim.x) {
        int n = i / 128, k = i % 128;
        Wt1[i] = (_Float16)W1[k * 16 + n];
    }
    for (int i = t; i < 16 * 160; i += blockDim.x) {
        int n = i / 160, k = i % 160;
        Wt2[i] = (_Float16)((k < 144) ? W2[k * 16 + n] : 0.0f);
        Wt3[i] = (_Float16)((k < 144) ? W3[k * 16 + n] : 0.0f);
        Wt4[i] = (_Float16)((n == 0 && k < 144) ? W4[k] : 0.0f);
    }
}

// ---------------------------------------------------------------------------
// Persistent-wave WMMA GEMM: each wave loops over 16-node tiles of
// h = [xa||x] @ W with B fragments (tile-invariant) hoisted into registers,
// then attention dots s = h*a_s, d = h*a_d via shfl_xor butterflies.
// Region select for the A fragment depends only on the chunk constant c,
// so it constant-folds after unrolling (no per-lane branches).
// ---------------------------------------------------------------------------
template <int KPAD, int HAS_XA>
__global__ void __launch_bounds__(256)
gat_gemm(const float* __restrict__ x, const float* __restrict__ xa,
         const _Float16* __restrict__ Wt, const float* __restrict__ avs,
         const float* __restrict__ avd, int alen,
         float* __restrict__ h, float* __restrict__ sarr, float* __restrict__ darr,
         int ntiles) {
    constexpr int NCH = KPAD / 32;
    const int lane = threadIdx.x & 31;
    const int n = lane & 15, half = lane >> 4;
    const int wave0  = (int)((blockIdx.x * blockDim.x + threadIdx.x) >> 5);
    const int nwaves = (int)((gridDim.x * blockDim.x) >> 5);

    // B fragments (32x16 f16 each): lane holds col N=n, K = {c+8*half..+7} and {c+16+8*half..+7}.
    v16h bf[NCH];
#pragma unroll
    for (int i = 0; i < NCH; ++i)
        bf[i] = __builtin_shufflevector(*(const v8h*)(Wt + (size_t)n * KPAD + i * 32 + 8 * half),
                                        *(const v8h*)(Wt + (size_t)n * KPAD + i * 32 + 16 + 8 * half),
                                        0, 1, 2, 3, 4, 5, 6, 7, 8, 9, 10, 11, 12, 13, 14, 15);

    const float asv = (n < alen) ? avs[n] : 0.0f;
    const float adv = (n < alen) ? avd[n] : 0.0f;

    for (int tile = wave0; tile < ntiles; tile += nwaves) {  // wave-uniform loop
        const int base = tile * 16;
        const float* xrow  = x + (size_t)(base + n) * 128 + 8 * half;
        const float* xarow = HAS_XA ? (xa + (size_t)(base + n) * 16 + 8 * half) : xrow;

        v8f acc = {};
#pragma unroll
        for (int i = 0; i < NCH; ++i) {
            const int c = i * 32;
            v8h lo, hi;   // A groups: virtual K {c+8*half..+7} and {c+16+8*half..+7}
            if (HAS_XA) {
                lo = (c == 0) ? cvt8(xarow) : cvt8(xrow + (c - 16));
                if (c + 16 >= 144) hi = (v8h)(_Float16)0.0f;   // zero-padded tail
                else               hi = cvt8(xrow + c);
            } else {
                lo = cvt8(xrow + c);
                hi = cvt8(xrow + c + 16);
            }
            v16h af = __builtin_shufflevector(lo, hi, 0, 1, 2, 3, 4, 5, 6, 7,
                                              8, 9, 10, 11, 12, 13, 14, 15);
            acc = __builtin_amdgcn_wmma_f32_16x16x32_f16(false, af, false, bf[i],
                                                         (short)0, acc, false, false);
        }

#pragma unroll
        for (int r = 0; r < 8; ++r) {        // acc[r] = h[base + r + 8*half][n]
            const int row = base + 8 * half + r;
            const float hv = acc[r];
            h[(size_t)row * 16 + n] = hv;
            float ts = hv * asv, td = hv * adv;
#pragma unroll
            for (int m = 1; m <= 8; m <<= 1) {   // butterfly within 16-lane half
                ts += __shfl_xor(ts, m, 32);
                td += __shfl_xor(td, m, 32);
            }
            if (n == 0) { sarr[row] = ts; darr[row] = td; }
        }
    }
}

// ---------------------------------------------------------------------------
// Segment-softmax passes over edges (+ implicit self-loops at e >= E).
// ---------------------------------------------------------------------------
__global__ void seg_init(float* __restrict__ emax, float* __restrict__ denom,
                         float* __restrict__ outacc, int N) {
    int i = blockIdx.x * blockDim.x + threadIdx.x;
    if (i >= N) return;
    emax[i] = -__builtin_inff();
    denom[i] = 0.0f;
    float4 z = make_float4(0.f, 0.f, 0.f, 0.f);
    float4* o = (float4*)(outacc + (size_t)i * 16);
    o[0] = z; o[1] = z; o[2] = z; o[3] = z;
}

__device__ __forceinline__ void edge_ids(int e, const int* src, const int* dst,
                                         int E, int& si, int& di) {
    if (e < E) { si = src[e]; di = dst[e]; }
    else       { si = e - E; di = si; }
}

__global__ void edge_max(const int* __restrict__ src, const int* __restrict__ dst,
                         const float* __restrict__ s, const float* __restrict__ d,
                         float* __restrict__ emax, int E, int N) {
    int e = blockIdx.x * blockDim.x + threadIdx.x;
    if (e >= E + N) return;
    int si, di; edge_ids(e, src, dst, E, si, di);
    atomicMaxF(emax + di, lrelu(s[si] + d[di]));
}

__global__ void edge_expsum(const int* __restrict__ src, const int* __restrict__ dst,
                            const float* __restrict__ s, const float* __restrict__ d,
                            const float* __restrict__ emax, float* __restrict__ denom,
                            int E, int N) {
    int e = blockIdx.x * blockDim.x + threadIdx.x;
    if (e >= E + N) return;
    int si, di; edge_ids(e, src, dst, E, si, di);
    atomicAdd(denom + di, __expf(lrelu(s[si] + d[di]) - emax[di]));
}

__global__ void edge_aggr16(const int* __restrict__ src, const int* __restrict__ dst,
                            const float* __restrict__ s, const float* __restrict__ d,
                            const float* __restrict__ emax, const float* __restrict__ denom,
                            const float* __restrict__ h, float* __restrict__ outacc,
                            int E, int N) {
    int e = blockIdx.x * blockDim.x + threadIdx.x;
    if (e >= E + N) return;
    int si, di; edge_ids(e, src, dst, E, si, di);
    float w = __expf(lrelu(s[si] + d[di]) - emax[di]) / denom[di];
    const float4* hr = (const float4*)(h + (size_t)si * 16);
    float4 h0 = hr[0], h1 = hr[1], h2 = hr[2], h3 = hr[3];
    float* o = outacc + (size_t)di * 16;
    atomicAdd(o + 0,  h0.x * w); atomicAdd(o + 1,  h0.y * w);
    atomicAdd(o + 2,  h0.z * w); atomicAdd(o + 3,  h0.w * w);
    atomicAdd(o + 4,  h1.x * w); atomicAdd(o + 5,  h1.y * w);
    atomicAdd(o + 6,  h1.z * w); atomicAdd(o + 7,  h1.w * w);
    atomicAdd(o + 8,  h2.x * w); atomicAdd(o + 9,  h2.y * w);
    atomicAdd(o + 10, h2.z * w); atomicAdd(o + 11, h2.w * w);
    atomicAdd(o + 12, h3.x * w); atomicAdd(o + 13, h3.y * w);
    atomicAdd(o + 14, h3.z * w); atomicAdd(o + 15, h3.w * w);
}

__global__ void edge_aggr1(const int* __restrict__ src, const int* __restrict__ dst,
                           const float* __restrict__ s, const float* __restrict__ d,
                           const float* __restrict__ emax, const float* __restrict__ denom,
                           const float* __restrict__ h, float* __restrict__ outacc,
                           int E, int N) {
    int e = blockIdx.x * blockDim.x + threadIdx.x;
    if (e >= E + N) return;
    int si, di; edge_ids(e, src, dst, E, si, di);
    float w = __expf(lrelu(s[si] + d[di]) - emax[di]) / denom[di];
    atomicAdd(outacc + di, h[(size_t)si * 16] * w);
}

__global__ void finalize16(const float* __restrict__ outacc, const float* __restrict__ b,
                           float* __restrict__ xa, int N) {
    int idx = blockIdx.x * blockDim.x + threadIdx.x;
    if (idx >= N * 16) return;
    xa[idx] = fmaxf(outacc[idx] + b[idx & 15], 0.0f);
}

__global__ void finalize1(const float* __restrict__ outacc, const float* __restrict__ b4,
                          float* __restrict__ px, int N) {
    int i = blockIdx.x * blockDim.x + threadIdx.x;
    if (i >= N) return;
    px[i] = fmaxf(outacc[i] + b4[0], 0.0f);
}

// ---------------------------------------------------------------------------
// Per-graph mean pool over layer-3 features + 16->16->1 MLP head, fused.
// ---------------------------------------------------------------------------
__global__ void __launch_bounds__(256)
pool_mlp(const float* __restrict__ xa, const float* __restrict__ fcw1,
         const float* __restrict__ fcb1, const float* __restrict__ fcw2,
         const float* __restrict__ fcb2, float* __restrict__ vx, int per_graph) {
    __shared__ float sm[256];
    __shared__ float hbuf[16];
    int g = blockIdx.x, t = threadIdx.x;
    int k = t & 15, chunk = t >> 4;
    float acc = 0.0f;
    for (int i = chunk; i < per_graph; i += 16)
        acc += xa[((size_t)g * per_graph + i) * 16 + k];
    sm[t] = acc;
    __syncthreads();
    for (int s = 128; s >= 16; s >>= 1) {    // reduce over chunk dim, keep k
        if (t < s) sm[t] += sm[t + s];
        __syncthreads();
    }
    if (t < 16) {
        float inv = 1.0f / (float)per_graph;
        float hj = fcb1[t];
        for (int kk = 0; kk < 16; ++kk)
            hj += (sm[kk] * inv) * fcw1[kk * 16 + t];
        hbuf[t] = fmaxf(hj, 0.0f);
    }
    __syncthreads();
    if (t == 0) {
        float r = fcb2[0];
        for (int j = 0; j < 16; ++j) r += hbuf[j] * fcw2[j];
        vx[g] = r;
    }
}

// ---------------------------------------------------------------------------
extern "C" void kernel_launch(void* const* d_in, const int* in_sizes, int n_in,
                              void* d_out, int out_size, void* d_ws, size_t ws_size,
                              hipStream_t stream) {
    const float* x   = (const float*)d_in[0];
    const int*   ei  = (const int*)d_in[1];
    // d_in[2] = edge_attr, unused by GATConv(edge_dim=None)
    const float* W1  = (const float*)d_in[3];
    const float* as1 = (const float*)d_in[4];
    const float* ad1 = (const float*)d_in[5];
    const float* b1  = (const float*)d_in[6];
    const float* W2  = (const float*)d_in[7];
    const float* as2 = (const float*)d_in[8];
    const float* ad2 = (const float*)d_in[9];
    const float* b2  = (const float*)d_in[10];
    const float* W3  = (const float*)d_in[11];
    const float* as3 = (const float*)d_in[12];
    const float* ad3 = (const float*)d_in[13];
    const float* b3  = (const float*)d_in[14];
    const float* W4  = (const float*)d_in[15];
    const float* as4 = (const float*)d_in[16];
    const float* ad4 = (const float*)d_in[17];
    const float* b4  = (const float*)d_in[18];
    const float* fcw1 = (const float*)d_in[19];
    const float* fcb1 = (const float*)d_in[20];
    const float* fcw2 = (const float*)d_in[21];
    const float* fcb2 = (const float*)d_in[22];

    const int N = in_sizes[0] / 128;
    const int E = in_sizes[1] / 2;
    const int* src = ei;
    const int* dst = ei + E;

    char* ws = (char*)d_ws;
    size_t o = 0;
    auto take = [&](size_t bytes) -> char* {
        char* p = ws + o;
        o += (bytes + 255) & ~(size_t)255;
        return p;
    };
    float*    h     = (float*)take((size_t)N * 16 * 4);
    float*    xa    = (float*)take((size_t)N * 16 * 4);
    float*    outac = (float*)take((size_t)N * 16 * 4);
    float*    sarr  = (float*)take((size_t)N * 4);
    float*    darr  = (float*)take((size_t)N * 4);
    float*    emax  = (float*)take((size_t)N * 4);
    float*    denom = (float*)take((size_t)N * 4);
    _Float16* Wt1   = (_Float16*)take(16 * 128 * 2);
    _Float16* Wt2   = (_Float16*)take(16 * 160 * 2);
    _Float16* Wt3   = (_Float16*)take(16 * 160 * 2);
    _Float16* Wt4   = (_Float16*)take(16 * 160 * 2);

    prep_weights<<<1, 256, 0, stream>>>(W1, W2, W3, W4, Wt1, Wt2, Wt3, Wt4);

    const int ntiles = N / 16;                       // N = 100000 -> 6250 tiles
    const int maxgb  = (ntiles * 32 + 255) / 256;    // one tile per wave upper bound
    const int ggrid  = maxgb < 192 ? maxgb : 192;    // persistent waves, ~4 tiles each
    const int egrid  = (E + N + 255) / 256;
    const int ngrid  = (N + 255) / 256;
    const int fgrid  = (N * 16 + 255) / 256;

    // ---- Layer 1: x @ W1 ----
    gat_gemm<128, 0><<<ggrid, 256, 0, stream>>>(x, nullptr, Wt1, as1, ad1, 16,
                                                h, sarr, darr, ntiles);
    seg_init<<<ngrid, 256, 0, stream>>>(emax, denom, outac, N);
    edge_max<<<egrid, 256, 0, stream>>>(src, dst, sarr, darr, emax, E, N);
    edge_expsum<<<egrid, 256, 0, stream>>>(src, dst, sarr, darr, emax, denom, E, N);
    edge_aggr16<<<egrid, 256, 0, stream>>>(src, dst, sarr, darr, emax, denom, h, outac, E, N);
    finalize16<<<fgrid, 256, 0, stream>>>(outac, b1, xa, N);

    // ---- Layer 2: [xa||x] @ W2 ----
    gat_gemm<160, 1><<<ggrid, 256, 0, stream>>>(x, xa, Wt2, as2, ad2, 16,
                                                h, sarr, darr, ntiles);
    seg_init<<<ngrid, 256, 0, stream>>>(emax, denom, outac, N);
    edge_max<<<egrid, 256, 0, stream>>>(src, dst, sarr, darr, emax, E, N);
    edge_expsum<<<egrid, 256, 0, stream>>>(src, dst, sarr, darr, emax, denom, E, N);
    edge_aggr16<<<egrid, 256, 0, stream>>>(src, dst, sarr, darr, emax, denom, h, outac, E, N);
    finalize16<<<fgrid, 256, 0, stream>>>(outac, b2, xa, N);

    // ---- Layer 3 ----
    gat_gemm<160, 1><<<ggrid, 256, 0, stream>>>(x, xa, Wt3, as3, ad3, 16,
                                                h, sarr, darr, ntiles);
    seg_init<<<ngrid, 256, 0, stream>>>(emax, denom, outac, N);
    edge_max<<<egrid, 256, 0, stream>>>(src, dst, sarr, darr, emax, E, N);
    edge_expsum<<<egrid, 256, 0, stream>>>(src, dst, sarr, darr, emax, denom, E, N);
    edge_aggr16<<<egrid, 256, 0, stream>>>(src, dst, sarr, darr, emax, denom, h, outac, E, N);
    finalize16<<<fgrid, 256, 0, stream>>>(outac, b3, xa, N);

    // ---- Layer 4: [xa||x] @ W4 (out dim 1, zero-padded to 16 cols) ----
    gat_gemm<160, 1><<<ggrid, 256, 0, stream>>>(x, xa, Wt4, as4, ad4, 1,
                                                h, sarr, darr, ntiles);
    seg_init<<<ngrid, 256, 0, stream>>>(emax, denom, outac, N);
    edge_max<<<egrid, 256, 0, stream>>>(src, dst, sarr, darr, emax, E, N);
    edge_expsum<<<egrid, 256, 0, stream>>>(src, dst, sarr, darr, emax, denom, E, N);
    edge_aggr1<<<egrid, 256, 0, stream>>>(src, dst, sarr, darr, emax, denom, h, outac, E, N);
    finalize1<<<ngrid, 256, 0, stream>>>(outac, b4, (float*)d_out, N);

    // ---- Mean pool (over layer-3 features) + MLP head ----
    pool_mlp<<<N / 1000, 256, 0, stream>>>(xa, fcw1, fcb1, fcw2, fcb2,
                                           (float*)d_out + N, 1000);
}
